// RelationalGCN_25082609008834
// MI455X (gfx1250) — matrix operand order repaired and checked by reading, not compiled, not run on hardware
//
#include <hip/hip_runtime.h>
#include <hip/hip_bf16.h>
#include <stdint.h>

#define N 4096
#define D 128
#define RREL 8
#define BBAS 8
#define MT (N/16)    // 256 m-tiles
#define KT (N/32)    // 128 k-steps of 32
#define NT (D/16)    // 8 n-tiles

typedef __attribute__((ext_vector_type(16))) __bf16 v16bf;
typedef __attribute__((ext_vector_type(16))) short v16s;
typedef __attribute__((ext_vector_type(16))) unsigned char v16u8;
typedef __attribute__((ext_vector_type(8)))  float v8f;

__device__ __forceinline__ unsigned short f2bf(float f) {
  union { float f; unsigned u; } c; c.f = f;
  unsigned u = c.u;
  u += 0x7fffu + ((u >> 16) & 1u);   // round-to-nearest-even
  return (unsigned short)(u >> 16);
}

// A-fragment (16-bit, 16x32): lane,element -> K = 16*(lane/16) + e (matches B-frag)
__device__ __forceinline__ int a_kof(int lane, int e) {
  int half = lane >> 4;
  int v = e >> 1, p = e & 1;
  int group = v >> 2, vv = v & 3;
  return half * 16 + e;
}

// -------- one-time: adj fp32 + et int -> bf16 A-fragments + u8 relation ids --------
__global__ void k_pack_adj(const float* __restrict__ adj, const int* __restrict__ et,
                           unsigned short* __restrict__ adjA, unsigned char* __restrict__ etA) {
  int gid = blockIdx.x * 256 + threadIdx.x;   // (mt, kt, lane)
  int lane = gid & 31;
  int kt = (gid >> 5) & (KT - 1);
  int mt = gid >> 12;
  int m = mt * 16 + (lane & 15);
  long base = (long)m * N + kt * 32;
  long oidx = ((long)(mt * KT + kt) * 32 + lane) * 16;
#pragma unroll
  for (int e = 0; e < 16; e++) {
    int K = a_kof(lane, e);
    adjA[oidx + e] = f2bf(adj[base + K]);
    etA[oidx + e] = (unsigned char)et[base + K];
  }
}

// -------- w[r] = sum_b weights[r,b] * bases[b]  (fp32, tiny) --------
__global__ void k_combine(const float* __restrict__ wts, const float* __restrict__ bases,
                          float* __restrict__ w) {
  int gid = blockIdx.x * 256 + threadIdx.x;   // 8*16384
  int r = gid >> 14;
  int io = gid & 16383;
  float acc = 0.f;
#pragma unroll
  for (int b = 0; b < BBAS; b++) acc += wts[r * BBAS + b] * bases[b * 16384 + io];
  w[gid] = acc;
}

// -------- xw[r] = h @ w[r] (r<8, bf16 B-fragment layout) ; r==8: hacc = h@W_self + b --------
// B-frag layout used here (32x16 bf16): n = lane%16, K = 16*(lane/16) + e  (contiguous k, same as A-frag)
__global__ void k_xw(const float* __restrict__ h, const float* __restrict__ w,
                     const float* __restrict__ wself, const float* __restrict__ bself,
                     unsigned short* __restrict__ xwfrag, float* __restrict__ hacc) {
  __shared__ float hs[64 * 128];
  int r = blockIdx.y;                 // 0..8
  int jb = blockIdx.x * 64;
  int t = threadIdx.x;
#pragma unroll
  for (int kk = 0; kk < 32; kk++) hs[t + kk * 256] = h[(long)jb * 128 + t + kk * 256];
  __syncthreads();
  const float* wmat = (r < 8) ? (w + r * 16384) : wself;
  int o = t & 127;
  int jl0 = (t >> 7) * 32;
  float acc[32];
#pragma unroll
  for (int j = 0; j < 32; j++) acc[j] = 0.f;
  for (int i = 0; i < 128; i++) {
    float wv = wmat[i * 128 + o];
#pragma unroll
    for (int j = 0; j < 32; j++) acc[j] += hs[(jl0 + j) * 128 + i] * wv;
  }
  if (r < 8) {
#pragma unroll
    for (int j = 0; j < 32; j++) {
      int jj = jb + jl0 + j;
      int kt = jj >> 5, K = jj & 31;
      int half = K >> 4, e = K & 15;
      int lane = (o & 15) + 16 * half;
      long idx = (((long)(r * KT + kt) * NT + (o >> 4)) * 32 + lane) * 16 + e;
      xwfrag[idx] = f2bf(acc[j]);
    }
  } else {
    float bb = bself[o];
#pragma unroll
    for (int j = 0; j < 32; j++) hacc[(long)(jb + jl0 + j) * 128 + o] = acc[j] + bb;
  }
}

// -------- aggregation: hacc += sum_r (adj . [et==r]) @ xw[r]  via bf16 WMMA --------
__global__ void __launch_bounds__(256) k_agg(const unsigned short* __restrict__ adjA,
                                             const unsigned char* __restrict__ etA,
                                             const unsigned short* __restrict__ xwfrag,
                                             float* __restrict__ hacc) {
  int lane = threadIdx.x & 31;
  int wave = threadIdx.x >> 5;
  int mt0 = blockIdx.x * 16 + wave * 2;     // 2 M-tiles per wave
  int kt0 = blockIdx.y * 16;                // split-K: 16 k-steps each

  // byte base pointers; all per-fragment deltas become immediate offsets (<8MB)
  const char* ap = (const char*)adjA  + ((long)(mt0 * KT + kt0) * 32 + lane) * 32;
  const char* ep = (const char*)etA   + ((long)(mt0 * KT + kt0) * 32 + lane) * 16;
  const char* bp = (const char*)xwfrag + ((long)kt0 * NT * 32 + lane) * 32;   // (r=0,n=0)

  const int MS_A = KT * 32 * 32;            // 131072: A byte stride between m-tiles
  const int MS_E = KT * 32 * 16;            // 65536
  // B strides: relation = KT*NT*32*32 = 1 MB, n-tile = 32*32 = 1 KB

  v8f acc[2][8] = {};
  for (int kt = 0; kt < 16; ++kt) {
    v16s a0 = *(const v16s*)(ap);
    v16s a1 = *(const v16s*)(ap + MS_A);
    v16u8 e0 = *(const v16u8*)(ep);
    v16u8 e1 = *(const v16u8*)(ep + MS_E);
    __builtin_prefetch(ap + 1024, 0, 1);
    __builtin_prefetch(ep + 512, 0, 1);
#pragma unroll
    for (int r = 0; r < 8; ++r) {
      v16s m0, m1;
#pragma unroll
      for (int e = 0; e < 16; ++e) {
        m0[e] = (e0[e] == r) ? a0[e] : (short)0;
        m1[e] = (e1[e] == r) ? a1[e] : (short)0;
      }
      v16bf am0 = __builtin_bit_cast(v16bf, m0);
      v16bf am1 = __builtin_bit_cast(v16bf, m1);
#pragma unroll
      for (int n = 0; n < 8; ++n) {
        v16s bs = *(const v16s*)(bp + r * 1048576 + n * 1024);
        v16bf bb = __builtin_bit_cast(v16bf, bs);
        acc[0][n] = __builtin_amdgcn_wmma_f32_16x16x32_bf16(false, am0, false, bb,
                                                            (short)0, acc[0][n], false, false);
        acc[1][n] = __builtin_amdgcn_wmma_f32_16x16x32_bf16(false, am1, false, bb,
                                                            (short)0, acc[1][n], false, false);
      }
    }
    ap += 1024; ep += 512; bp += 8192;
  }

  int row_in = (lane >> 4) * 8;             // C layout: M = v + 8*(lane/16), N = lane%16
  int col = lane & 15;
#pragma unroll
  for (int m = 0; m < 2; m++) {
    int mbase = (mt0 + m) * 16 + row_in;
#pragma unroll
    for (int n = 0; n < 8; n++)
#pragma unroll
      for (int v = 0; v < 8; v++)
        unsafeAtomicAdd(&hacc[(long)(mbase + v) * 128 + n * 16 + col], acc[m][n][v]);
  }
}

// -------- layernorm (+optional relu), one wave32 per row --------
__global__ void k_ln(const float* __restrict__ hin, const float* __restrict__ gamma,
                     const float* __restrict__ beta, float* __restrict__ out, int relu) {
  int lane = threadIdx.x & 31;
  int wid = threadIdx.x >> 5;
  int row = blockIdx.x * 8 + wid;
  const float* p = hin + (long)row * 128;
  float v[4], s = 0.f, sq = 0.f;
#pragma unroll
  for (int q = 0; q < 4; q++) { v[q] = p[lane + 32 * q]; s += v[q]; sq += v[q] * v[q]; }
#pragma unroll
  for (int off = 16; off > 0; off >>= 1) { s += __shfl_xor(s, off, 32); sq += __shfl_xor(sq, off, 32); }
  float mu = s * (1.f / 128.f);
  float var = sq * (1.f / 128.f) - mu * mu;
  float rs = rsqrtf(var + 1e-5f);
  float* po = out + (long)row * 128;
#pragma unroll
  for (int q = 0; q < 4; q++) {
    int i = lane + 32 * q;
    float y = (v[q] - mu) * rs * gamma[i] + beta[i];
    if (relu) y = fmaxf(y, 0.f);
    po[i] = y;
  }
}

extern "C" void kernel_launch(void* const* d_in, const int* in_sizes, int n_in,
                              void* d_out, int out_size, void* d_ws, size_t ws_size,
                              hipStream_t stream) {
  const float* x      = (const float*)d_in[0];
  const float* adj    = (const float*)d_in[1];
  const int*   et     = (const int*)d_in[2];
  const float* bases1 = (const float*)d_in[3];
  const float* wts1   = (const float*)d_in[4];
  const float* Wself1 = (const float*)d_in[5];
  const float* bself1 = (const float*)d_in[6];
  const float* bases2 = (const float*)d_in[7];
  const float* wts2   = (const float*)d_in[8];
  const float* Wself2 = (const float*)d_in[9];
  const float* bself2 = (const float*)d_in[10];
  const float* g1 = (const float*)d_in[11];
  const float* b1 = (const float*)d_in[12];
  const float* g2 = (const float*)d_in[13];
  const float* b2 = (const float*)d_in[14];

  char* ws = (char*)d_ws;
  unsigned short* adjA   = (unsigned short*)ws;                           // 32 MB
  unsigned char*  etA    = (unsigned char*)(ws + 32ull * 1024 * 1024);    // 16 MB
  float*          wbuf   = (float*)(ws + 48ull * 1024 * 1024);            // 512 KB
  unsigned short* xwfrag = (unsigned short*)(ws + 49ull * 1024 * 1024);   // 8 MB
  float*          hacc   = (float*)(ws + 57ull * 1024 * 1024);            // 2 MB
  float*          h1     = (float*)(ws + 59ull * 1024 * 1024);            // 2 MB

  k_pack_adj<<<4096, 256, 0, stream>>>(adj, et, adjA, etA);
  // layer 1
  k_combine<<<512, 256, 0, stream>>>(wts1, bases1, wbuf);
  k_xw<<<dim3(64, 9), 256, 0, stream>>>(x, wbuf, Wself1, bself1, xwfrag, hacc);
  k_agg<<<dim3(16, 8), 256, 0, stream>>>(adjA, etA, xwfrag, hacc);
  k_ln<<<512, 256, 0, stream>>>(hacc, g1, b1, h1, 1);
  // layer 2
  k_combine<<<512, 256, 0, stream>>>(wts2, bases2, wbuf);
  k_xw<<<dim3(64, 9), 256, 0, stream>>>(h1, wbuf, Wself2, bself2, xwfrag, hacc);
  k_agg<<<dim3(16, 8), 256, 0, stream>>>(adjA, etA, xwfrag, hacc);
  k_ln<<<512, 256, 0, stream>>>(hacc, g2, b2, (float*)d_out, 0);
}
